// MultiHeadSelfAttention_82111184764940
// MI455X (gfx1250) — matrix operand order repaired and checked by reading, not compiled, not run on hardware
//
#include <hip/hip_runtime.h>

// ---------------------------------------------------------------------------
// MHA forward for MI455X (gfx1250), bf16 WMMA with f32 accumulation.
// B=2, T=2048, D=1024, H=16, DK=64 (hardcoded).
// ---------------------------------------------------------------------------

typedef __attribute__((ext_vector_type(16))) __bf16 v16bf;
typedef __attribute__((ext_vector_type(8)))  float  v8f;
typedef __attribute__((ext_vector_type(4)))  int    v4i;

#if defined(__has_builtin)
#if __has_builtin(__builtin_amdgcn_global_load_async_to_lds_b128) && \
    __has_builtin(__builtin_amdgcn_s_wait_asynccnt)
#define USE_ASYNC_LDS 1
#endif
#endif
#ifndef USE_ASYNC_LDS
#define USE_ASYNC_LDS 0
#endif

#if USE_ASYNC_LDS
typedef __attribute__((address_space(1))) v4i* gptr_v4i;
typedef __attribute__((address_space(3))) v4i* lptr_v4i;
#endif

// Copy 16 bytes global -> LDS. Async (ASYNCcnt-tracked) when available.
__device__ __forceinline__ void copy16(const __bf16* g, __bf16* l) {
#if USE_ASYNC_LDS
  __builtin_amdgcn_global_load_async_to_lds_b128((gptr_v4i)g, (lptr_v4i)l, 0, 0);
#else
  *(uint4*)l = *(const uint4*)g;
#endif
}

__device__ __forceinline__ void wait_async() {
#if USE_ASYNC_LDS
  __builtin_amdgcn_s_wait_asynccnt(0);
#endif
}

__device__ __forceinline__ v8f wmma_bf16(v16bf a, v16bf b, v8f c) {
  return __builtin_amdgcn_wmma_f32_16x16x32_bf16(
      /*neg_a=*/false, a, /*neg_b=*/false, b,
      /*c_mod=*/(short)0, c, /*reuse_a=*/false, /*reuse_b=*/false);
}

// A-matrix fragment, 16x32 bf16 (MxK). ISA 7.12.2:
// lanes 0-15: M=lane, K-base 0; lanes 16-31: M=lane-16, K-base 8.
__device__ __forceinline__ v16bf load_a_frag(const __bf16* tile, int stride, int lane) {
  const int m  = lane & 15;
  const int kb = (lane >> 4) << 3;
  const __bf16* row = tile + m * stride + kb;
  v16bf a;
#pragma unroll
  for (int p = 0; p < 8; ++p) {
    const int k = (p < 4) ? (2 * p) : (8 + 2 * p);  // p>=4 -> 16 + 2(p-4)
    a[2 * p]     = row[k];
    a[2 * p + 1] = row[k + 1];
  }
  return a;
}

// B-matrix fragment, 32x16 bf16 (KxN), tile stored [n][k] row-major.
// lanes 0-15: N=lane, K-base 0; lanes 16-31: N=lane-16, K-base 8 (same K order as A).
__device__ __forceinline__ v16bf load_b_frag(const __bf16* tile, int stride, int lane) {
  const int n  = lane & 15;
  const int kb = (lane >> 4) << 3;
  const __bf16* col = tile + n * stride + kb;
  v16bf b;
#pragma unroll
  for (int p = 0; p < 8; ++p) {
    const int k = (p < 4) ? (2 * p) : (8 + 2 * p);
    b[2 * p]     = col[k];
    b[2 * p + 1] = col[k + 1];
  }
  return b;
}

// ---------------------------------------------------------------------------
__global__ __launch_bounds__(256) void f32_to_bf16(const float* __restrict__ in,
                                                   __bf16* __restrict__ out, int n) {
  const int i = blockIdx.x * 256 + threadIdx.x;
  if (i < n) out[i] = (__bf16)in[i];
}

// ---------------------------------------------------------------------------
// C[n,m] = A[n,:] . W[m,:] + bias[m]  (x @ W^T + b), K = 1024.
// Block tile 128x128, 8 waves: wave w -> rows (w&3)*32 (2 frags),
// cols (w>>2)*64 (4 frags) => 8 WMMA per K-step. Double-buffered async fills.
// mode 0: out bf16 scattered to [B,H,T,DK] (value scaled by `scale`)
// mode 1: out f32 row-major [N,1024]
__global__ __launch_bounds__(256) void gemm_bf16_nt(
    const __bf16* __restrict__ A, const __bf16* __restrict__ W,
    const float* __restrict__ bias, void* __restrict__ outp,
    int mode, float scale) {
  __shared__ __bf16 sA[2][128][40];   // stride 80B: 16B-aligned rows
  __shared__ __bf16 sB[2][128][40];
  const int tid  = threadIdx.x;
  const int wave = tid >> 5, lane = tid & 31;
  const int rowblk = (wave & 3) << 5;   // 0,32,64,96
  const int colblk = (wave >> 2) << 6;  // 0,64
  const int nbase  = blockIdx.y << 7;
  const int mbase  = blockIdx.x << 7;

  v8f acc[2][4] = {};

  // fill one K-slab (128 rows x 32 k for A and B): 1024 octs of 16B
  auto fill = [&](int buf, int kk) {
#pragma unroll
    for (int it = 0; it < 2; ++it) {
      const int q  = tid + (it << 8);   // 512 octs per matrix
      const int r  = q >> 2;            // 128 rows, 4 octs/row
      const int k8 = (q & 3) << 3;
      copy16(A + (size_t)(nbase + r) * 1024 + kk + k8, &sA[buf][r][k8]);
      copy16(W + (size_t)(mbase + r) * 1024 + kk + k8, &sB[buf][r][k8]);
    }
  };

  fill(0, 0);
  int buf = 0;
  for (int kk = 0; kk < 1024; kk += 32, buf ^= 1) {
    wait_async();
    __syncthreads();                    // current buf ready on all waves
    if (kk + 32 < 1024) fill(buf ^ 1, kk + 32);  // overlap DMA with compute
    v16bf a0 = load_a_frag(&sA[buf][rowblk][0], 40, lane);
    v16bf a1 = load_a_frag(&sA[buf][rowblk + 16][0], 40, lane);
#pragma unroll
    for (int j = 0; j < 4; ++j) {
      v16bf b = load_b_frag(&sB[buf][colblk + (j << 4)][0], 40, lane);
      acc[0][j] = wmma_bf16(a0, b, acc[0][j]);
      acc[1][j] = wmma_bf16(a1, b, acc[1][j]);
    }
  }

  const int rowoff = (lane >> 4) << 3;
  const int ncol   = lane & 15;
#pragma unroll
  for (int i = 0; i < 2; ++i) {
#pragma unroll
    for (int j = 0; j < 4; ++j) {
      const int gm = mbase + colblk + (j << 4) + ncol;
      const float bv = bias[gm];
#pragma unroll
      for (int r = 0; r < 8; ++r) {
        const int gn = nbase + rowblk + (i << 4) + rowoff + r;
        const float v = (acc[i][j][r] + bv) * scale;
        if (mode == 0) {
          const int h = gm >> 6, dk = gm & 63;
          const int bb = gn >> 11, tt = gn & 2047;
          ((__bf16*)outp)[(((size_t)bb * 16 + h) * 2048 + tt) * 64 + dk] = (__bf16)v;
        } else {
          ((float*)outp)[(size_t)gn * 1024 + gm] = v;
        }
      }
    }
  }
}

// ---------------------------------------------------------------------------
// Causal flash attention. Q pre-scaled by 1/8. Q,K,V: bf16 [B*H, T, 64].
// Grid (T/128, B*H). Block 256 = 8 waves; wave owns 16 query rows.
__global__ __launch_bounds__(256) void attn_fwd(
    const __bf16* __restrict__ Q, const __bf16* __restrict__ K,
    const __bf16* __restrict__ V, __bf16* __restrict__ O) {
  __shared__ __bf16 sK[64][72];        // [key][d], 144B rows (16B aligned)
  __shared__ __bf16 sVt[64][72];       // [d][key]
  __shared__ __bf16 sP[8][16][68];     // per-wave P staging [qrow][key]

  const int tid  = threadIdx.x;
  const int wave = tid >> 5, lane = tid & 31;
  const int bh   = blockIdx.y;
  const int qbase = (blockIdx.x << 7) + (wave << 4);
  const size_t slab = (size_t)bh * 2048 * 64;

  const __bf16* Qs = Q + slab + (size_t)qbase * 64;
  const v16bf qa0 = load_a_frag(Qs, 64, lane);        // d 0..31
  const v16bf qa1 = load_a_frag(Qs + 32, 64, lane);   // d 32..63

  float rm[8], rs[8];
#pragma unroll
  for (int r = 0; r < 8; ++r) { rm[r] = -1e30f; rs[r] = 0.f; }
  v8f acc[4] = {};

  const int mrow = (lane >> 4) << 3;
  const int ncol = lane & 15;
  __bf16* sPw = &sP[wave][0][0];

  const int ktmax = ((blockIdx.x << 7) + 127) >> 6;
  for (int kt = 0; kt <= ktmax; ++kt) {
    const int kb0 = kt << 6;
    const __bf16* Ks = K + slab + (size_t)kb0 * 64;
    const __bf16* Vs = V + slab + (size_t)kb0 * 64;
    // K tile: async DMA into LDS (512 octs)
#pragma unroll
    for (int it = 0; it < 2; ++it) {
      const int q  = tid + (it << 8);
      const int r  = q >> 3;            // 64 rows, 8 octs/row
      const int d8 = (q & 7) << 3;
      copy16(Ks + r * 64 + d8, &sK[r][d8]);
    }
    // V tile: sync load + transpose into [d][key]
#pragma unroll
    for (int it = 0; it < 4; ++it) {
      const int q4  = tid + (it << 8);   // 1024 quads over 64x64 tile
      const int row = q4 >> 4;
      const int d4  = (q4 & 15) << 2;
      __bf16 tmp[4];
      *(uint2*)tmp = *(const uint2*)(Vs + row * 64 + d4);
#pragma unroll
      for (int j = 0; j < 4; ++j) sVt[d4 + j][row] = tmp[j];
    }
    wait_async();
    __syncthreads();

    if (kb0 <= qbase + 15) {             // wave has unmasked keys in this tile
      v8f s[4];
#pragma unroll
      for (int t = 0; t < 4; ++t) {
        v16bf b0 = load_b_frag(&sK[t << 4][0], 72, lane);
        v16bf b1 = load_b_frag(&sK[t << 4][32], 72, lane);
        v8f sc = {};
        sc = wmma_bf16(qa0, b0, sc);
        sc = wmma_bf16(qa1, b1, sc);
        s[t] = sc;
      }
      // causal mask + online softmax (row = vgpr index within 16-lane half)
#pragma unroll
      for (int r = 0; r < 8; ++r) {
        const int qrow = qbase + mrow + r;
        float mx = -1e30f;
#pragma unroll
        for (int t = 0; t < 4; ++t) {
          const int key = kb0 + (t << 4) + ncol;
          float v = s[t][r];
          v = (key <= qrow) ? v : -1e30f;
          s[t][r] = v;
          mx = fmaxf(mx, v);
        }
#pragma unroll
        for (int off = 8; off > 0; off >>= 1)
          mx = fmaxf(mx, __shfl_xor(mx, off, 32));
        const float newm  = fmaxf(rm[r], mx);
        const float alpha = __expf(rm[r] - newm);
        float rsum = 0.f;
#pragma unroll
        for (int t = 0; t < 4; ++t) {
          const float e = __expf(s[t][r] - newm);
          rsum += e;
          sPw[(mrow + r) * 68 + (t << 4) + ncol] = (__bf16)e;
        }
#pragma unroll
        for (int off = 8; off > 0; off >>= 1)
          rsum += __shfl_xor(rsum, off, 32);
        rs[r] = rs[r] * alpha + rsum;
        rm[r] = newm;
#pragma unroll
        for (int t = 0; t < 4; ++t) acc[t][r] *= alpha;
      }
      // P @ V (wave-private LDS staging; DS ops in-order within a wave)
#pragma unroll
      for (int ks = 0; ks < 2; ++ks) {
        v16bf pa = load_a_frag(sPw + (ks << 5), 68, lane);
#pragma unroll
        for (int t = 0; t < 4; ++t) {
          v16bf vb = load_b_frag(&sVt[t << 4][ks << 5], 72, lane);
          acc[t] = wmma_bf16(pa, vb, acc[t]);
        }
      }
    }
    __syncthreads();
  }

  // epilogue: normalize, write bf16 [B,T,D] (head-interleaved row-major)
  const int b = bh >> 4, h = bh & 15;
#pragma unroll
  for (int t = 0; t < 4; ++t) {
#pragma unroll
    for (int r = 0; r < 8; ++r) {
      const int qrow = qbase + mrow + r;
      const int d = (t << 4) + ncol;
      const float o = acc[t][r] / rs[r];
      O[((size_t)(b * 2048 + qrow)) * 1024 + h * 64 + d] = (__bf16)o;
    }
  }
}

// ---------------------------------------------------------------------------
extern "C" void kernel_launch(void* const* d_in, const int* in_sizes, int n_in,
                              void* d_out, int out_size, void* d_ws, size_t ws_size,
                              hipStream_t stream) {
  (void)in_sizes; (void)n_in; (void)out_size; (void)ws_size;
  const float* x  = (const float*)d_in[0];
  const float* wq = (const float*)d_in[1];
  const float* bq = (const float*)d_in[2];
  const float* wk = (const float*)d_in[3];
  const float* bk = (const float*)d_in[4];
  const float* wv = (const float*)d_in[5];
  const float* bv = (const float*)d_in[6];
  const float* wo = (const float*)d_in[7];
  const float* bo = (const float*)d_in[8];

  char* ws = (char*)d_ws;
  const size_t MB = 1u << 20;
  __bf16* xb  = (__bf16*)(ws + 0 * MB);    // 4096x1024  (8 MB)
  __bf16* wqb = (__bf16*)(ws + 8 * MB);    // 1024x1024  (2 MB each)
  __bf16* wkb = (__bf16*)(ws + 10 * MB);
  __bf16* wvb = (__bf16*)(ws + 12 * MB);
  __bf16* wob = (__bf16*)(ws + 14 * MB);
  __bf16* Qb  = (__bf16*)(ws + 16 * MB);   // [B*H, T, 64] (8 MB each)
  __bf16* Kb  = (__bf16*)(ws + 24 * MB);
  __bf16* Vb  = (__bf16*)(ws + 32 * MB);
  __bf16* Ab  = (__bf16*)(ws + 40 * MB);   // attn out [B,T,D] (8 MB)

  const int nx = 4096 * 1024, nw = 1024 * 1024;
  f32_to_bf16<<<nx / 256, 256, 0, stream>>>(x,  xb,  nx);
  f32_to_bf16<<<nw / 256, 256, 0, stream>>>(wq, wqb, nw);
  f32_to_bf16<<<nw / 256, 256, 0, stream>>>(wk, wkb, nw);
  f32_to_bf16<<<nw / 256, 256, 0, stream>>>(wv, wvb, nw);
  f32_to_bf16<<<nw / 256, 256, 0, stream>>>(wo, wob, nw);

  dim3 gg(8, 32), bb256(256);              // 128x128 tiles over 1024x4096
  // Q gets the 1/sqrt(DK)=0.125 attention scale folded in.
  gemm_bf16_nt<<<gg, bb256, 0, stream>>>(xb, wqb, bq, (void*)Qb, 0, 0.125f);
  gemm_bf16_nt<<<gg, bb256, 0, stream>>>(xb, wkb, bk, (void*)Kb, 0, 1.0f);
  gemm_bf16_nt<<<gg, bb256, 0, stream>>>(xb, wvb, bv, (void*)Vb, 0, 1.0f);

  attn_fwd<<<dim3(16, 32), bb256, 0, stream>>>(Qb, Kb, Vb, Ab);

  gemm_bf16_nt<<<gg, bb256, 0, stream>>>(Ab, wob, bo, d_out, 1, 1.0f);
}
